// TransformerLayer_26233660244513
// MI455X (gfx1250) — matrix operand order repaired and checked by reading, not compiled, not run on hardware
//
#include <hip/hip_runtime.h>
#include <hip/hip_bf16.h>
#include <hip/hip_fp16.h>

typedef __attribute__((ext_vector_type(16))) _Float16 v16h;
typedef __attribute__((ext_vector_type(8)))  _Float16 v8h;
typedef __attribute__((ext_vector_type(8)))  float    v8f;

#define HEADS 8
#define DH    64
#define LM    256
#define NSEQ  8192
#define BATCH 4
#define DIM   512
#define BH    (BATCH*HEADS)   // 32
#define BN    (BATCH*NSEQ)    // 32768

// ---------------------------------------------------------------- helpers

__device__ __forceinline__ v8f zero8() {
  v8f z;
#pragma unroll
  for (int i = 0; i < 8; ++i) z[i] = 0.f;
  return z;
}

__device__ __forceinline__ v8f wmma_f16(v16h a, v16h b, v8f c) {
  return __builtin_amdgcn_wmma_f32_16x16x32_f16(false, a, false, b, (short)0, c,
                                                false, false);
}

// A fragment: tile stored row-major [16 rows(M)][>=32 cols(K)], ld multiple of 8,
// base 16B aligned. Lane L: row = L&15; k = 16*(L>>4) + {0..15}.
__device__ __forceinline__ v16h load_a_frag(const _Float16* t, int ld) {
  int lane = threadIdx.x & 31;
  const _Float16* r = t + (size_t)(lane & 15) * ld + ((lane >> 4) << 4);
  v8h lo = *(const v8h*)r;
  v8h hi = *(const v8h*)(r + 8);
  v16h f;
#pragma unroll
  for (int e = 0; e < 8; ++e) { f[e] = lo[e]; f[8 + e] = hi[e]; }
  return f;
}

// Swizzled B panel: slot layout [tile][lane][16 halves] (tile = nt*ktiles + kt).
// Lane L, element e holds B[kt*32 + (L&16) + e][nt*16 + (L&15)].
__device__ __forceinline__ int bswz_off(int kk, int nn, int ktiles) {
  return (((nn >> 4) * ktiles + (kk >> 5)) * 32 + ((kk & 16) | (nn & 15))) * 16 +
         (kk & 15);
}

// Global panel-swizzled layout for a (K x N) matrix consumed in 32-row k-steps
// and 64-col n-panels: panel id = (kk/32)*nblks + nn/64, 2048 halves per panel.
__device__ __forceinline__ size_t gswz_off(int kk, int nn, int nblks) {
  return (size_t)((kk >> 5) * nblks + (nn >> 6)) * 2048 +
         (size_t)bswz_off(kk & 31, nn & 63, 1);
}

__device__ __forceinline__ v16h load_b_swz(const _Float16* panel, int tile) {
  int lane = threadIdx.x & 31;
  const _Float16* p = panel + ((size_t)(tile * 32 + lane) << 4);
  v8h lo = *(const v8h*)p;
  v8h hi = *(const v8h*)(p + 8);
  v16h f;
#pragma unroll
  for (int e = 0; e < 8; ++e) { f[e] = lo[e]; f[8 + e] = hi[e]; }
  return f;
}

// ------------------------- CDNA5 async global->LDS copy (16 bytes / lane) ----

typedef __attribute__((address_space(3))) _Float16* lds_hp;

__device__ __forceinline__ void async_b128(_Float16* lds_dst, const _Float16* gsrc) {
#if defined(__gfx1250__)
  unsigned loff = (unsigned)(unsigned long long)(lds_hp)lds_dst;
  unsigned long long ga = (unsigned long long)gsrc;
  asm volatile("global_load_async_to_lds_b128 %0, %1, off"
               :: "v"(loff), "v"(ga)
               : "memory");
#else
#pragma unroll
  for (int i = 0; i < 8; ++i) lds_dst[i] = gsrc[i];
#endif
}

__device__ __forceinline__ void async_wait0() {
#if defined(__gfx1250__)
  asm volatile("s_wait_asynccnt 0" ::: "memory");
#endif
}

// ---------------------------------------------------------------- LayerNorm

__global__ void ln_kernel(const float* __restrict__ x, const float* __restrict__ g,
                          const float* __restrict__ be, _Float16* __restrict__ xn) {
  int row  = blockIdx.x * 8 + (threadIdx.x >> 5);
  int lane = threadIdx.x & 31;
  const float* xr = x + (size_t)row * DIM;
  float v[16];
  float s = 0.f;
#pragma unroll
  for (int i = 0; i < 16; ++i) { v[i] = xr[i*32 + lane]; s += v[i]; }
#pragma unroll
  for (int off = 16; off > 0; off >>= 1) s += __shfl_xor(s, off, 32);
  float mu = s * (1.0f / DIM);
  float s2 = 0.f;
#pragma unroll
  for (int i = 0; i < 16; ++i) { float d = v[i] - mu; s2 += d * d; }
#pragma unroll
  for (int off = 16; off > 0; off >>= 1) s2 += __shfl_xor(s2, off, 32);
  float rstd = rsqrtf(s2 * (1.0f / DIM) + 1e-5f);
  _Float16* o = xn + (size_t)row * DIM;
#pragma unroll
  for (int i = 0; i < 16; ++i) {
    int c = i*32 + lane;
    o[c] = (_Float16)((v[i] - mu) * rstd * g[c] + be[c]);
  }
}

// f32 (KxN row-major) -> f16 global panel-swizzled (one-time weight conversion)
__global__ void cvt_swz(const float* __restrict__ s, _Float16* __restrict__ d,
                        int N, int nblks) {
  int idx = blockIdx.x * blockDim.x + threadIdx.x;
  int kk = idx / N, nn = idx - kk * N;
  d[gswz_off(kk, nn, nblks)] = (_Float16)s[idx];
}

// v (b,h,n,d) -> per-64-token-chunk swizzled panels (K=64 tokens, N=64 dh)
__global__ void vswz_kernel(const _Float16* __restrict__ v, _Float16* __restrict__ vs) {
  size_t idx = (size_t)blockIdx.x * blockDim.x + threadIdx.x;
  int d  = (int)(idx & 63);
  int n  = (int)((idx >> 6) & 8191);
  int bh = (int)(idx >> 19);
  int chunk = n >> 6, r = n & 63;
  vs[((size_t)bh * 128 + chunk) * 4096 + bswz_off(r, d, 2)] = v[idx];
}

// ---------------------------------------------------------------- QKV GEMM

__global__ void qkv_gemm(const _Float16* __restrict__ xn, const _Float16* __restrict__ wsw,
                         _Float16* __restrict__ q, _Float16* __restrict__ k,
                         _Float16* __restrict__ v) {
  __shared__ __attribute__((aligned(16))) _Float16 As[128][40];
  __shared__ __attribute__((aligned(16))) _Float16 Bs[4 * 32 * 16];
  int t = threadIdx.x, wave = t >> 5, lane = t & 31;
  int m0 = blockIdx.x * 128;
  int n0 = blockIdx.y * 64;
  v8f acc[4];
#pragma unroll
  for (int c = 0; c < 4; ++c) acc[c] = zero8();
  for (int k0 = 0; k0 < DIM; k0 += 32) {
    __syncthreads();
    { // A tile 128x32 (async)
      int r = t >> 1, seg = (t & 1) * 16;
      const _Float16* src = xn + (size_t)(m0 + r) * DIM + k0 + seg;
      async_b128(&As[r][seg], src);
      async_b128(&As[r][seg + 8], src + 8);
    }
    { // B panel: pre-swizzled weights -> single async per thread
      const _Float16* src =
          wsw + ((size_t)((k0 >> 5) * 24 + blockIdx.y)) * 2048 + t * 8;
      async_b128(&Bs[t * 8], src);
    }
    async_wait0();
    __syncthreads();
    v16h a = load_a_frag(&As[wave * 16][0], 40);
    v16h bf[4];
#pragma unroll
    for (int c = 0; c < 4; ++c) bf[c] = load_b_swz(Bs, c);
#pragma unroll
    for (int c = 0; c < 4; ++c) acc[c] = wmma_f16(a, bf[c], acc[c]);
  }
  int col = lane & 15, rbase = (lane >> 4) << 3;
#pragma unroll
  for (int c = 0; c < 4; ++c) {
    int gcol = n0 + c * 16 + col;
    int part = gcol >> 9;            // 0=q 1=k 2=v
    int rem  = gcol & 511;
    int hh = rem >> 6, dd = rem & 63;
    _Float16* dst = (part == 0) ? q : (part == 1) ? k : v;
    float scale = (part == 0) ? 0.125f : 1.0f;   // dh^-0.5
#pragma unroll
    for (int i = 0; i < 8; ++i) {
      int grow = m0 + wave * 16 + rbase + i;
      int bb = grow >> 13, nn = grow & 8191;
      dst[((size_t)(bb * HEADS + hh) * NSEQ + nn) * DH + dd] =
          (_Float16)(acc[c][i] * scale);
    }
  }
}

// ---------------------------------------------------------------- landmarks

__global__ void landmark_kernel(const _Float16* __restrict__ q,
                                const _Float16* __restrict__ k,
                                _Float16* __restrict__ ql, _Float16* __restrict__ kl) {
  int bh = blockIdx.x >> 8;
  int mm = blockIdx.x & 255;
  int d  = threadIdx.x;   // 64 threads
  size_t base = ((size_t)bh * NSEQ + (size_t)mm * 32) * DH + d;
  float sq = 0.f, sk = 0.f;
#pragma unroll
  for (int tk = 0; tk < 32; ++tk) {
    sq += (float)q[base + (size_t)tk * DH];
    sk += (float)k[base + (size_t)tk * DH];
  }
  size_t ob = ((size_t)bh * LM + mm) * DH + d;
  ql[ob] = (_Float16)(sq * (1.f / 32.f));
  kl[ob] = (_Float16)(sk * (1.f / 32.f));
}

// ---------------------------------------------------------------- attn2 = softmax(ql @ kl^T)

__global__ void attn2_kernel(const _Float16* __restrict__ ql,
                             const _Float16* __restrict__ kl,
                             _Float16* __restrict__ a2) {
  __shared__ __attribute__((aligned(16))) _Float16 As[128][72];
  __shared__ __attribute__((aligned(16))) _Float16 Bp[16 * 2 * 32 * 16];
  int t = threadIdx.x, wave = t >> 5, lane = t & 31;
  int bh = blockIdx.y;
  int r0 = blockIdx.x * 128;
  { int r = t >> 1, seg = (t & 1) * 32;
    const _Float16* src = ql + ((size_t)bh * LM + r0 + r) * DH + seg;
#pragma unroll
    for (int u = 0; u < 4; ++u) async_b128(&As[r][seg + u * 8], src + u * 8);
  }
  { // kl^T panel (K=64, N=256): contiguous both sides -> async
    int j = t, nt = j >> 4, nl = j & 15;
    const _Float16* src = kl + ((size_t)bh * LM + j) * DH;
#pragma unroll
    for (int kt = 0; kt < 2; ++kt)
#pragma unroll
      for (int hi = 0; hi < 2; ++hi) {
        _Float16* dst = Bp + (size_t)(((nt * 2 + kt) * 32) + hi * 16 + nl) * 16;
        const _Float16* s2 = src + kt * 32 + hi * 16;
        async_b128(dst, s2);
        async_b128(dst + 8, s2 + 8);
      }
  }
  async_wait0();
  __syncthreads();
  v16h aq0 = load_a_frag(&As[wave * 16][0], 72);
  v16h aq1 = load_a_frag(&As[wave * 16][32], 72);
  v8f s[16];
#pragma unroll
  for (int ct = 0; ct < 16; ++ct) {
    v8f a0 = zero8();
    a0 = wmma_f16(aq0, load_b_swz(Bp, ct * 2 + 0), a0);
    a0 = wmma_f16(aq1, load_b_swz(Bp, ct * 2 + 1), a0);
    s[ct] = a0;
  }
  float m[8], li[8];
#pragma unroll
  for (int i = 0; i < 8; ++i) {
    float mx = -1e30f;
#pragma unroll
    for (int ct = 0; ct < 16; ++ct) mx = fmaxf(mx, s[ct][i]);
#pragma unroll
    for (int off = 8; off > 0; off >>= 1) mx = fmaxf(mx, __shfl_xor(mx, off, 32));
    m[i] = mx;
  }
#pragma unroll
  for (int i = 0; i < 8; ++i) {
    float sum = 0.f;
#pragma unroll
    for (int ct = 0; ct < 16; ++ct) {
      float e = __expf(s[ct][i] - m[i]); s[ct][i] = e; sum += e;
    }
#pragma unroll
    for (int off = 8; off > 0; off >>= 1) sum += __shfl_xor(sum, off, 32);
    li[i] = 1.0f / sum;
  }
  int col = lane & 15, rbase = (lane >> 4) << 3;
#pragma unroll
  for (int ct = 0; ct < 16; ++ct)
#pragma unroll
    for (int i = 0; i < 8; ++i) {
      int grow = r0 + wave * 16 + rbase + i;
      a2[((size_t)bh * LM + grow) * LM + ct * 16 + col] = (_Float16)(s[ct][i] * li[i]);
    }
}

// ----------- global max of column sums (rows sum to 1 -> col factor == 1)

__global__ void init_colmax(unsigned* colmax) { if (threadIdx.x == 0) *colmax = 0u; }

__global__ void colmax_kernel(const _Float16* __restrict__ a2, unsigned* colmax) {
  int bh = blockIdx.x, col = threadIdx.x;
  float s = 0.f;
  for (int j = 0; j < LM; ++j)
    s += (float)a2[((size_t)bh * LM + j) * LM + col];
  atomicMax(colmax, __float_as_uint(s));  // nonneg floats: uint order == float order
}

// z0 = a^T / rowmax, written row-major (A-operand) and swizzled (B-operand)
__global__ void zinit_kernel(const _Float16* __restrict__ a2, const unsigned* colmax,
                             _Float16* __restrict__ zrm, _Float16* __restrict__ zsw) {
  int idx = blockIdx.x * blockDim.x + threadIdx.x;
  int bh = idx >> 16, rem = idx & 65535, i = rem >> 8, j = rem & 255;
  float rmax = __uint_as_float(*colmax);
  float val = (float)a2[((size_t)bh * LM + j) * LM + i] / rmax;
  zrm[idx] = (_Float16)val;
  zsw[(size_t)bh * 65536 + gswz_off(i, j, 4)] = (_Float16)val;
}

// y_sw = dg*I - x   (consumed only as B operand -> swizzled output)
__global__ void ifma_kernel(const _Float16* __restrict__ x, _Float16* __restrict__ ysw,
                            float dg) {
  int idx = blockIdx.x * blockDim.x + threadIdx.x;
  int bh = idx >> 16, rem = idx & 65535, i = rem >> 8, j = rem & 255;
  float val = ((i == j) ? dg : 0.f) - (float)x[idx];
  ysw[(size_t)bh * 65536 + gswz_off(i, j, 4)] = (_Float16)val;
}

// ---------------------------------------------------------------- batched 256^3 GEMM (pinv)
// A row-major, B pre-swizzled; C row-major (+ optional swizzled copy for B-reuse)

__global__ void gemm_bh256(const _Float16* __restrict__ A, const _Float16* __restrict__ Bsw,
                           _Float16* __restrict__ C, _Float16* __restrict__ Csw,
                           float alpha) {
  __shared__ __attribute__((aligned(16))) _Float16 As[128][40];
  __shared__ __attribute__((aligned(16))) _Float16 Bs[4 * 32 * 16];
  int t = threadIdx.x, wave = t >> 5, lane = t & 31;
  int bh = blockIdx.z;
  int m0 = blockIdx.x * 128, n0 = blockIdx.y * 64;
  const _Float16* Ab = A + (size_t)bh * LM * LM;
  const _Float16* Bb = Bsw + (size_t)bh * LM * LM;
  v8f acc[4];
#pragma unroll
  for (int c = 0; c < 4; ++c) acc[c] = zero8();
  for (int k0 = 0; k0 < LM; k0 += 32) {
    __syncthreads();
    { int r = t >> 1, seg = (t & 1) * 16;
      const _Float16* src = Ab + (size_t)(m0 + r) * LM + k0 + seg;
      async_b128(&As[r][seg], src);
      async_b128(&As[r][seg + 8], src + 8); }
    { const _Float16* src =
          Bb + ((size_t)((k0 >> 5) * 4 + (n0 >> 6))) * 2048 + t * 8;
      async_b128(&Bs[t * 8], src); }
    async_wait0();
    __syncthreads();
    v16h a = load_a_frag(&As[wave * 16][0], 40);
    v16h bf[4];
#pragma unroll
    for (int c = 0; c < 4; ++c) bf[c] = load_b_swz(Bs, c);
#pragma unroll
    for (int c = 0; c < 4; ++c) acc[c] = wmma_f16(a, bf[c], acc[c]);
  }
  int col = lane & 15, rbase = (lane >> 4) << 3;
  _Float16* Cb = C + (size_t)bh * LM * LM;
#pragma unroll
  for (int c = 0; c < 4; ++c)
#pragma unroll
    for (int i = 0; i < 8; ++i)
      Cb[(size_t)(m0 + wave * 16 + rbase + i) * LM + n0 + c * 16 + col] =
          (_Float16)(acc[c][i] * alpha);
  if (Csw) {
    _Float16* Cs = Csw + (size_t)bh * 65536;
#pragma unroll
    for (int c = 0; c < 4; ++c)
#pragma unroll
      for (int i = 0; i < 8; ++i)
        Cs[gswz_off(m0 + wave * 16 + rbase + i, n0 + c * 16 + col, 4)] =
            (_Float16)(acc[c][i] * alpha);
  }
}

// ------------------------------------------- flash attn3 @ v : kv3 = softmax(ql@k^T) @ v

__global__ void attn3v_kernel(const _Float16* __restrict__ ql,
                              const _Float16* __restrict__ kk,
                              const _Float16* __restrict__ vsw,
                              float* __restrict__ kv3) {
  __shared__ __attribute__((aligned(16))) _Float16 Kp[4 * 2 * 32 * 16];
  __shared__ __attribute__((aligned(16))) _Float16 Vp[4 * 2 * 32 * 16];
  __shared__ __attribute__((aligned(16))) _Float16 Pw[8][16][72];
  int t = threadIdx.x, wave = t >> 5, lane = t & 31;
  int bh = blockIdx.y;
  int r0 = blockIdx.x * 128 + wave * 16;
  const _Float16* aptr = ql + ((size_t)bh * LM + r0) * DH;
  v16h aq0 = load_a_frag(aptr, DH);
  v16h aq1 = load_a_frag(aptr + 32, DH);
  float m[8], l[8];
#pragma unroll
  for (int i = 0; i < 8; ++i) { m[i] = -1e30f; l[i] = 0.f; }

  // Pass A: streaming row max / denominator
  for (int c = 0; c < NSEQ / 64; ++c) {
    { int j = t & 63, g = t >> 6;
      int kt = g >> 1, hi = g & 1;
      int nt = j >> 4, nl = j & 15;
      const _Float16* src =
          kk + ((size_t)bh * NSEQ + c * 64 + j) * DH + kt * 32 + hi * 16;
      _Float16* dst = Kp + (size_t)(((nt * 2 + kt) * 32) + hi * 16 + nl) * 16;
      async_b128(dst, src);
      async_b128(dst + 8, src + 8);
    }
    async_wait0();
    __syncthreads();
    v8f s[4];
#pragma unroll
    for (int ct = 0; ct < 4; ++ct) {
      v8f a0 = zero8();
      a0 = wmma_f16(aq0, load_b_swz(Kp, ct * 2 + 0), a0);
      a0 = wmma_f16(aq1, load_b_swz(Kp, ct * 2 + 1), a0);
      s[ct] = a0;
    }
#pragma unroll
    for (int i = 0; i < 8; ++i) {
      float mx = -1e30f;
#pragma unroll
      for (int ct = 0; ct < 4; ++ct) mx = fmaxf(mx, s[ct][i]);
#pragma unroll
      for (int off = 8; off > 0; off >>= 1) mx = fmaxf(mx, __shfl_xor(mx, off, 32));
      float mn = fmaxf(m[i], mx);
      float sum = 0.f;
#pragma unroll
      for (int ct = 0; ct < 4; ++ct) sum += __expf(s[ct][i] - mn);
#pragma unroll
      for (int off = 8; off > 0; off >>= 1) sum += __shfl_xor(sum, off, 32);
      l[i] = l[i] * __expf(m[i] - mn) + sum;
      m[i] = mn;
    }
    __syncthreads();
  }

  // Pass B: O += exp(S - m) @ V
  v8f o[4];
#pragma unroll
  for (int ct = 0; ct < 4; ++ct) o[ct] = zero8();
  int col = lane & 15, rbase = (lane >> 4) << 3;
  for (int c = 0; c < NSEQ / 64; ++c) {
    { int j = t & 63, g = t >> 6;
      int kt = g >> 1, hi = g & 1;
      int nt = j >> 4, nl = j & 15;
      const _Float16* src =
          kk + ((size_t)bh * NSEQ + c * 64 + j) * DH + kt * 32 + hi * 16;
      _Float16* dst = Kp + (size_t)(((nt * 2 + kt) * 32) + hi * 16 + nl) * 16;
      async_b128(dst, src);
      async_b128(dst + 8, src + 8); }
    { // V panel already swizzled in global -> pure async
      const _Float16* src = vsw + ((size_t)bh * 128 + c) * 4096 + t * 16;
      async_b128(&Vp[t * 16], src);
      async_b128(&Vp[t * 16 + 8], src + 8); }
    async_wait0();
    __syncthreads();
#pragma unroll
    for (int ct = 0; ct < 4; ++ct) {
      v8f a0 = zero8();
      a0 = wmma_f16(aq0, load_b_swz(Kp, ct * 2 + 0), a0);
      a0 = wmma_f16(aq1, load_b_swz(Kp, ct * 2 + 1), a0);
#pragma unroll
      for (int i = 0; i < 8; ++i)
        Pw[wave][rbase + i][ct * 16 + col] = (_Float16)__expf(a0[i] - m[i]);
    }
    __syncthreads();
#pragma unroll
    for (int ks = 0; ks < 2; ++ks) {
      v16h a = load_a_frag(&Pw[wave][0][ks * 32], 72);
#pragma unroll
      for (int ct = 0; ct < 4; ++ct)
        o[ct] = wmma_f16(a, load_b_swz(Vp, ct * 2 + ks), o[ct]);
    }
    __syncthreads();
  }
#pragma unroll
  for (int ct = 0; ct < 4; ++ct)
#pragma unroll
    for (int i = 0; i < 8; ++i)
      kv3[((size_t)bh * LM + r0 + rbase + i) * DH + ct * 16 + col] = o[ct][i] / l[i];
}

// ---------------------------------------------------------------- y = pinv @ kv3 (swizzled out)

__global__ void ygemm_kernel(const _Float16* __restrict__ Z, const float* __restrict__ KV,
                             _Float16* __restrict__ Ysw) {
  __shared__ __attribute__((aligned(16))) _Float16 As[128][40];
  __shared__ __attribute__((aligned(16))) _Float16 Bs[4 * 32 * 16];
  int t = threadIdx.x, wave = t >> 5, lane = t & 31;
  int bh = blockIdx.z;
  int m0 = blockIdx.x * 128;
  const _Float16* Ab = Z + (size_t)bh * LM * LM;
  const float* Bb = KV + (size_t)bh * LM * DH;
  v8f acc[4];
#pragma unroll
  for (int c = 0; c < 4; ++c) acc[c] = zero8();
  for (int k0 = 0; k0 < LM; k0 += 32) {
    __syncthreads();
    { int r = t >> 1, seg = (t & 1) * 16;
      const _Float16* src = Ab + (size_t)(m0 + r) * LM + k0 + seg;
      async_b128(&As[r][seg], src);
      async_b128(&As[r][seg + 8], src + 8); }
    { int r = t >> 3, seg = (t & 7) * 8;
      const float* src = Bb + (size_t)(k0 + r) * DH + seg;
#pragma unroll
      for (int j = 0; j < 8; ++j) Bs[bswz_off(r, seg + j, 1)] = (_Float16)src[j]; }
    async_wait0();
    __syncthreads();
    v16h a = load_a_frag(&As[wave * 16][0], 40);
    v16h bf[4];
#pragma unroll
    for (int c = 0; c < 4; ++c) bf[c] = load_b_swz(Bs, c);
#pragma unroll
    for (int c = 0; c < 4; ++c) acc[c] = wmma_f16(a, bf[c], acc[c]);
  }
  int col = lane & 15, rbase = (lane >> 4) << 3;
  _Float16* Yb = Ysw + (size_t)bh * LM * DH;
#pragma unroll
  for (int c = 0; c < 4; ++c)
#pragma unroll
    for (int i = 0; i < 8; ++i)
      Yb[bswz_off(m0 + wave * 16 + rbase + i, c * 16 + col, 8)] = (_Float16)acc[c][i];
}

// ------------------------------ out_heads = softmax(q@kl^T) @ y  (fused)

__global__ void attn1_kernel(const _Float16* __restrict__ q,
                             const _Float16* __restrict__ kl,
                             const _Float16* __restrict__ ysw,
                             float* __restrict__ outh) {
  __shared__ __attribute__((aligned(16))) _Float16 As[128][72];
  __shared__ __attribute__((aligned(16))) _Float16 Big[16384]; // kl^T panel / y panel
  __shared__ __attribute__((aligned(16))) _Float16 Pw[8][16][40];
  int t = threadIdx.x, wave = t >> 5, lane = t & 31;
  int bh = blockIdx.y;
  int n0 = blockIdx.x * 128;
  { int r = t >> 1, seg = (t & 1) * 32;
    const _Float16* src = q + ((size_t)bh * NSEQ + n0 + r) * DH + seg;
#pragma unroll
    for (int u = 0; u < 4; ++u) async_b128(&As[r][seg + u * 8], src + u * 8); }
  { int j = t, nt = j >> 4, nl = j & 15;
    const _Float16* src = kl + ((size_t)bh * LM + j) * DH;
#pragma unroll
    for (int kt = 0; kt < 2; ++kt)
#pragma unroll
      for (int hi = 0; hi < 2; ++hi) {
        _Float16* dst = Big + (size_t)(((nt * 2 + kt) * 32) + hi * 16 + nl) * 16;
        const _Float16* s2 = src + kt * 32 + hi * 16;
        async_b128(dst, s2);
        async_b128(dst + 8, s2 + 8);
      }
  }
  async_wait0();
  __syncthreads();
  v16h aq0 = load_a_frag(&As[wave * 16][0], 72);
  v16h aq1 = load_a_frag(&As[wave * 16][32], 72);
  v8f s[16];
#pragma unroll
  for (int ct = 0; ct < 16; ++ct) {
    v8f a0 = zero8();
    a0 = wmma_f16(aq0, load_b_swz(Big, ct * 2 + 0), a0);
    a0 = wmma_f16(aq1, load_b_swz(Big, ct * 2 + 1), a0);
    s[ct] = a0;
  }
  float m[8], li[8];
#pragma unroll
  for (int i = 0; i < 8; ++i) {
    float mx = -1e30f;
#pragma unroll
    for (int ct = 0; ct < 16; ++ct) mx = fmaxf(mx, s[ct][i]);
#pragma unroll
    for (int off = 8; off > 0; off >>= 1) mx = fmaxf(mx, __shfl_xor(mx, off, 32));
    m[i] = mx;
  }
#pragma unroll
  for (int i = 0; i < 8; ++i) {
    float sum = 0.f;
#pragma unroll
    for (int ct = 0; ct < 16; ++ct) {
      float e = __expf(s[ct][i] - m[i]); s[ct][i] = e; sum += e;
    }
#pragma unroll
    for (int off = 8; off > 0; off >>= 1) sum += __shfl_xor(sum, off, 32);
    li[i] = 1.0f / sum;
  }
#pragma unroll
  for (int ct = 0; ct < 16; ++ct)
#pragma unroll
    for (int i = 0; i < 8; ++i) s[ct][i] *= li[i];
  __syncthreads();
  { // restage Big as y panel from pre-swizzled global -> pure async
    const _Float16* src = ysw + (size_t)bh * 16384 + t * 64;
#pragma unroll
    for (int u = 0; u < 8; ++u) async_b128(&Big[t * 64 + u * 8], src + u * 8);
  }
  async_wait0();
  __syncthreads();
  v8f o[4];
#pragma unroll
  for (int c = 0; c < 4; ++c) o[c] = zero8();
  int col = lane & 15, rbase = (lane >> 4) << 3;
  for (int kc = 0; kc < 8; ++kc) {
#pragma unroll
    for (int tt = 0; tt < 2; ++tt)
#pragma unroll
      for (int i = 0; i < 8; ++i)
        Pw[wave][rbase + i][tt * 16 + col] = (_Float16)(s[kc * 2 + tt][i]);
    __syncthreads();
    v16h a = load_a_frag(&Pw[wave][0][0], 40);
#pragma unroll
    for (int ct = 0; ct < 4; ++ct)
      o[ct] = wmma_f16(a, load_b_swz(Big, ct * 8 + kc), o[ct]);
    __syncthreads();
  }
#pragma unroll
  for (int ct = 0; ct < 4; ++ct)
#pragma unroll
    for (int i = 0; i < 8; ++i)
      outh[((size_t)bh * NSEQ + n0 + wave * 16 + rbase + i) * DH + ct * 16 + col] =
          o[ct][i];
}

// ---------------------------------------------------------------- depthwise conv residual

__global__ void conv_kernel(const _Float16* __restrict__ v, const float* __restrict__ rk,
                            float* __restrict__ outh) {
  size_t idx = (size_t)blockIdx.x * blockDim.x + threadIdx.x;
  int d  = (int)(idx & 63);
  int n  = (int)((idx >> 6) & 8191);
  int bh = (int)(idx >> 19);
  int h  = bh & 7;
  const _Float16* vb = v + (size_t)bh * NSEQ * DH + d;
  float acc = 0.f;
#pragma unroll
  for (int u = 0; u < 33; ++u) {
    int nn = n + u - 16;
    if (nn >= 0 && nn < NSEQ) acc += (float)vb[(size_t)nn * DH] * rk[h * 33 + u];
  }
  outh[idx] += acc;
}

// ---------------------------------------------------------------- output projection + residual

__global__ void outproj_kernel(const float* __restrict__ outh,
                               const _Float16* __restrict__ wosw,
                               const float* __restrict__ bo,
                               const float* __restrict__ x,
                               float* __restrict__ out) {
  __shared__ __attribute__((aligned(16))) _Float16 As[128][40];
  __shared__ __attribute__((aligned(16))) _Float16 Bs[4 * 32 * 16];
  int t = threadIdx.x, wave = t >> 5, lane = t & 31;
  int m0 = blockIdx.x * 128, n0 = blockIdx.y * 64;
  v8f acc[4];
#pragma unroll
  for (int c = 0; c < 4; ++c) acc[c] = zero8();
  for (int k0 = 0; k0 < DIM; k0 += 32) {
    int h = k0 >> 6, d0 = k0 & 63;
    __syncthreads();
    { int r = t >> 1, seg = (t & 1) * 16;
      int gr = m0 + r, bb = gr >> 13, nn = gr & 8191;
      const float* src = outh + ((size_t)(bb * HEADS + h) * NSEQ + nn) * DH + d0 + seg;
#pragma unroll
      for (int j = 0; j < 16; ++j) As[r][seg + j] = (_Float16)src[j]; }
    { const _Float16* src =
          wosw + ((size_t)((k0 >> 5) * 8 + blockIdx.y)) * 2048 + t * 8;
      async_b128(&Bs[t * 8], src); }
    async_wait0();
    __syncthreads();
    v16h a = load_a_frag(&As[wave * 16][0], 40);
    v16h bf[4];
#pragma unroll
    for (int c = 0; c < 4; ++c) bf[c] = load_b_swz(Bs, c);
#pragma unroll
    for (int c = 0; c < 4; ++c) acc[c] = wmma_f16(a, bf[c], acc[c]);
  }
  int col = lane & 15, rbase = (lane >> 4) << 3;
#pragma unroll
  for (int c = 0; c < 4; ++c) {
    int gcol = n0 + c * 16 + col;
#pragma unroll
    for (int i = 0; i < 8; ++i) {
      int grow = m0 + wave * 16 + rbase + i;
      out[(size_t)grow * DIM + gcol] =
          x[(size_t)grow * DIM + gcol] + acc[c][i] + bo[gcol];
    }
  }
}

// ---------------------------------------------------------------- launch

extern "C" void kernel_launch(void* const* d_in, const int* in_sizes, int n_in,
                              void* d_out, int out_size, void* d_ws, size_t ws_size,
                              hipStream_t stream) {
  (void)in_sizes; (void)n_in; (void)out_size; (void)ws_size;
  const float* x    = (const float*)d_in[0];
  const float* lg   = (const float*)d_in[1];
  const float* lb   = (const float*)d_in[2];
  const float* wqkv = (const float*)d_in[3];
  const float* wout = (const float*)d_in[4];
  const float* bout = (const float*)d_in[5];
  const float* rk   = (const float*)d_in[6];
  float* out = (float*)d_out;

  char* wsb = (char*)d_ws;
  size_t off = 0;
  auto alloc = [&](size_t bytes) -> void* {
    void* p = wsb + off;
    off = (off + bytes + 255) & ~(size_t)255;
    return p;
  };
  _Float16* xn16  = (_Float16*)alloc((size_t)BN * DIM * 2);
  _Float16* wsw   = (_Float16*)alloc((size_t)DIM * 1536 * 2);
  _Float16* wosw  = (_Float16*)alloc((size_t)DIM * DIM * 2);
  _Float16* q16   = (_Float16*)alloc((size_t)BH * NSEQ * DH * 2);
  _Float16* k16   = (_Float16*)alloc((size_t)BH * NSEQ * DH * 2);
  _Float16* v16   = (_Float16*)alloc((size_t)BH * NSEQ * DH * 2);
  _Float16* vsw   = (_Float16*)alloc((size_t)BH * NSEQ * DH * 2);
  _Float16* ql16  = (_Float16*)alloc((size_t)BH * LM * DH * 2);
  _Float16* kl16  = (_Float16*)alloc((size_t)BH * LM * DH * 2);
  _Float16* a2    = (_Float16*)alloc((size_t)BH * LM * LM * 2);
  _Float16* zArm  = (_Float16*)alloc((size_t)BH * LM * LM * 2);
  _Float16* zBrm  = (_Float16*)alloc((size_t)BH * LM * LM * 2);
  _Float16* zAsw  = (_Float16*)alloc((size_t)BH * LM * LM * 2);
  _Float16* zBsw  = (_Float16*)alloc((size_t)BH * LM * LM * 2);
  _Float16* az    = (_Float16*)alloc((size_t)BH * LM * LM * 2);
  _Float16* tbsw  = (_Float16*)alloc((size_t)BH * LM * LM * 2);
  _Float16* ub    = (_Float16*)alloc((size_t)BH * LM * LM * 2);
  unsigned* colmax = (unsigned*)alloc(256);
  float* kv3   = (float*)alloc((size_t)BH * LM * DH * 4);
  _Float16* ysw = (_Float16*)alloc((size_t)BH * LM * DH * 2);
  float* outh  = (float*)alloc((size_t)BH * NSEQ * DH * 4);

  init_colmax<<<1, 32, 0, stream>>>(colmax);
  cvt_swz<<<(DIM * 1536) / 256, 256, 0, stream>>>(wqkv, wsw, 1536, 24);
  cvt_swz<<<(DIM * DIM) / 256, 256, 0, stream>>>(wout, wosw, DIM, 8);
  ln_kernel<<<BN / 8, 256, 0, stream>>>(x, lg, lb, xn16);
  qkv_gemm<<<dim3(BN / 128, 24), 256, 0, stream>>>(xn16, wsw, q16, k16, v16);
  vswz_kernel<<<(unsigned)((size_t)BH * NSEQ * DH / 256), 256, 0, stream>>>(v16, vsw);
  landmark_kernel<<<BH * LM, 64, 0, stream>>>(q16, k16, ql16, kl16);
  attn2_kernel<<<dim3(2, BH), 256, 0, stream>>>(ql16, kl16, a2);
  colmax_kernel<<<BH, LM, 0, stream>>>(a2, colmax);
  zinit_kernel<<<(BH * LM * LM) / 256, 256, 0, stream>>>(a2, colmax, zArm, zAsw);

  _Float16* zcrm = zArm;
  _Float16* zcsw = zAsw;
  _Float16* znrm = zBrm;
  _Float16* znsw = zBsw;
  const int elemBlocks = (BH * LM * LM) / 256;
  for (int it = 0; it < 6; ++it) {
    gemm_bh256<<<dim3(2, 4, BH), 256, 0, stream>>>(a2, zcsw, az, (_Float16*)nullptr, 1.0f);
    ifma_kernel<<<elemBlocks, 256, 0, stream>>>(az, tbsw, 7.0f);
    gemm_bh256<<<dim3(2, 4, BH), 256, 0, stream>>>(az, tbsw, ub, (_Float16*)nullptr, 1.0f);
    ifma_kernel<<<elemBlocks, 256, 0, stream>>>(ub, tbsw, 15.0f);
    gemm_bh256<<<dim3(2, 4, BH), 256, 0, stream>>>(az, tbsw, ub, (_Float16*)nullptr, 1.0f);
    ifma_kernel<<<elemBlocks, 256, 0, stream>>>(ub, tbsw, 13.0f);
    gemm_bh256<<<dim3(2, 4, BH), 256, 0, stream>>>(zcrm, tbsw, znrm, znsw, 0.25f);
    _Float16* tr = zcrm; zcrm = znrm; znrm = tr;
    _Float16* ts = zcsw; zcsw = znsw; znsw = ts;
  }

  attn3v_kernel<<<dim3(2, BH), 256, 0, stream>>>(ql16, k16, vsw, kv3);
  ygemm_kernel<<<dim3(2, 1, BH), 256, 0, stream>>>(zcrm, kv3, ysw);
  attn1_kernel<<<dim3(NSEQ / 128, BH), 256, 0, stream>>>(q16, kl16, ysw, outh);
  conv_kernel<<<(unsigned)((size_t)BH * NSEQ * DH / 256), 256, 0, stream>>>(v16, rk, outh);
  outproj_kernel<<<dim3(BN / 128, 8), 256, 0, stream>>>(outh, wosw, bout, x, out);
}
